// GCNEncoder_88149908783548
// MI455X (gfx1250) — compile-verified, matrix operand order checked by reading, not run on hardware
//
#include <hip/hip_runtime.h>

typedef __attribute__((ext_vector_type(16))) __bf16 v16bf;
typedef __attribute__((ext_vector_type(8)))  __bf16 v8bf;
typedef __attribute__((ext_vector_type(4)))  __bf16 v4bf;
typedef __attribute__((ext_vector_type(8)))  float  v8f;

#define IN_F  128
#define HID_F 128
#define OUT_F 64

// ---------------- utility kernels ----------------

__global__ void k_fill_f32(float* __restrict__ p, int n, float v) {
    int i = blockIdx.x * blockDim.x + threadIdx.x;
    if (i < n) p[i] = v;
}

__global__ void k_zero_f32(float* __restrict__ p, int n) {
    int i = blockIdx.x * blockDim.x + threadIdx.x;
    if (i < n) p[i] = 0.0f;
}

__global__ void k_deg_count(const int* __restrict__ dst, float* __restrict__ deg, int E) {
    int e = blockIdx.x * blockDim.x + threadIdx.x;
    if (e < E) atomicAdd(&deg[dst[e]], 1.0f);
}

__global__ void k_rsqrt(float* __restrict__ d, int n) {
    int i = blockIdx.x * blockDim.x + threadIdx.x;
    if (i < n) {
        float v = d[i];
        d[i] = v > 0.0f ? __frsqrt_rn(v) : 0.0f;
    }
}

__global__ void k_cvt_f32_bf16(const float* __restrict__ in, __bf16* __restrict__ out, int n) {
    int i = blockIdx.x * blockDim.x + threadIdx.x;
    if (i < n) out[i] = (__bf16)in[i];
}

// ---------------- weight pre-pack into WMMA B-operand register layout ----------------
// Wp[((tn*KB + kb)*32 + lane)*16 + j] = bf16( W[k*Nc + col] )
//   col = tn*16 + (lane&15), half = lane>>4
//   k   = kb*32 + (j<8 ? half*8 + j : 16 + half*8 + (j-8))
// GEMM then reads each lane's 16 B-values as two contiguous b128 loads.
__global__ void k_pack_w(const float* __restrict__ W, __bf16* __restrict__ Wp,
                         int K, int Nc) {
    int idx = blockIdx.x * blockDim.x + threadIdx.x;
    if (idx >= K * Nc) return;
    int j    = idx & 15;
    int lane = (idx >> 4) & 31;
    int rest = idx >> 9;          // tn*KB + kb
    int KB   = K >> 5;
    int kb   = rest % KB;
    int tn   = rest / KB;
    int half = lane >> 4;
    int col  = tn * 16 + (lane & 15);
    int k    = kb * 32 + ((j < 8) ? (half * 8 + j) : (16 + half * 8 + (j - 8)));
    Wp[idx] = (__bf16)W[k * Nc + col];
}

// ---------------- bf16 WMMA GEMM: C[M,NC] = A[M,K] * W[K,NC] (W pre-packed) --------
// One wave computes 16 rows x NC cols: A loaded once per K-block, reused for all
// NC/16 column tiles. Fully unrolled -> KB*TN v_wmma_f32_16x16x32_bf16 per wave.
template <int K, int NC>
__global__ void k_gemm_wmma_bf16(const __bf16* __restrict__ A,
                                 const __bf16* __restrict__ Wp,
                                 float* __restrict__ C, int M) {
    constexpr int KB = K / 32;
    constexpr int TN = NC / 16;

    int wave = threadIdx.x >> 5;
    int lane = threadIdx.x & 31;
    int rowTile = blockIdx.x * (blockDim.x >> 5) + wave;
    if (rowTile * 16 >= M) return;            // wave-uniform: EXEC stays all-ones

    int half = lane >> 4;
    int row  = rowTile * 16 + (lane & 15);
    const __bf16* arow = A + (size_t)row * K;
    const __bf16* blane = Wp + (size_t)lane * 16;

    v8f acc[TN] = {};

#pragma unroll
    for (int kb = 0; kb < KB; ++kb) {
        v8bf alo = *(const v8bf*)(arow + kb * 32 + half * 8);
        v8bf ahi = *(const v8bf*)(arow + kb * 32 + 16 + half * 8);
        v16bf a = __builtin_shufflevector(alo, ahi,
                      0, 1, 2, 3, 4, 5, 6, 7, 8, 9, 10, 11, 12, 13, 14, 15);
#pragma unroll
        for (int tn = 0; tn < TN; ++tn) {
            const __bf16* bp = blane + (size_t)(tn * KB + kb) * 32 * 16;
            v8bf blo = *(const v8bf*)(bp);
            v8bf bhi = *(const v8bf*)(bp + 8);
            v16bf b = __builtin_shufflevector(blo, bhi,
                          0, 1, 2, 3, 4, 5, 6, 7, 8, 9, 10, 11, 12, 13, 14, 15);
            acc[tn] = __builtin_amdgcn_wmma_f32_16x16x32_bf16(
                          false, a, false, b, (short)0, acc[tn], false, false);
        }
    }

    // C/D layout: VGPR r -> row (r + 8*half), col = tn*16 + (lane&15)
    int crow = rowTile * 16 + half * 8;
#pragma unroll
    for (int tn = 0; tn < TN; ++tn) {
        int col = tn * 16 + (lane & 15);
#pragma unroll
        for (int r = 0; r < 8; ++r)
            C[(size_t)(crow + r) * NC + col] = acc[tn][r];
    }
}

// ---------------- edge aggregation: one wave per edge ----------------
// agg[dst] += hlin[src] * dinv[src]*dinv[dst]  (atomics stay in the 192MB L2)
__global__ void k_scatter(const float* __restrict__ hlin,
                          const int* __restrict__ src,
                          const int* __restrict__ dst,
                          const float* __restrict__ dinv,
                          float* __restrict__ agg, int E, int F) {
    int wave = (blockIdx.x * blockDim.x + threadIdx.x) >> 5;
    int lane = threadIdx.x & 31;
    if (wave >= E) return;
    int s = src[wave];
    int d = dst[wave];
    float nrm = dinv[s] * dinv[d];
    const float* hs = hlin + (size_t)s * F;
    float* ad = agg + (size_t)d * F;
    for (int f = lane; f < F; f += 32)
        atomicAdd(&ad[f], hs[f] * nrm);
}

// ---------------- finalize: + self-loop + bias (+relu), vectorized x4 ----------------
__global__ void k_finalize_bf16(const float* __restrict__ agg,
                                const float* __restrict__ hlin,
                                const float* __restrict__ dinv,
                                const float* __restrict__ bias,
                                __bf16* __restrict__ out, int N, int F) {
    int g = blockIdx.x * blockDim.x + threadIdx.x;     // group of 4 features
    int total = (N * F) >> 2;
    if (g >= total) return;
    int per = F >> 2;
    int i  = g / per;
    int fg = g - i * per;
    float di = dinv[i];
    float dd = di * di;
    float4 a = ((const float4*)agg)[g];
    float4 h = ((const float4*)hlin)[g];
    float4 b = ((const float4*)bias)[fg];
    float v0 = a.x + h.x * dd + b.x;
    float v1 = a.y + h.y * dd + b.y;
    float v2 = a.z + h.z * dd + b.z;
    float v3 = a.w + h.w * dd + b.w;
    v4bf o;
    o[0] = (__bf16)(v0 > 0.0f ? v0 : 0.0f);
    o[1] = (__bf16)(v1 > 0.0f ? v1 : 0.0f);
    o[2] = (__bf16)(v2 > 0.0f ? v2 : 0.0f);
    o[3] = (__bf16)(v3 > 0.0f ? v3 : 0.0f);
    ((v4bf*)out)[g] = o;
}

__global__ void k_finalize_f32(const float* __restrict__ agg,
                               const float* __restrict__ hlin,
                               const float* __restrict__ dinv,
                               const float* __restrict__ bias,
                               float* __restrict__ out, int N, int F) {
    int g = blockIdx.x * blockDim.x + threadIdx.x;
    int total = (N * F) >> 2;
    if (g >= total) return;
    int per = F >> 2;
    int i  = g / per;
    int fg = g - i * per;
    float di = dinv[i];
    float dd = di * di;
    float4 a = ((const float4*)agg)[g];
    float4 h = ((const float4*)hlin)[g];
    float4 b = ((const float4*)bias)[fg];
    float4 o;
    o.x = a.x + h.x * dd + b.x;
    o.y = a.y + h.y * dd + b.y;
    o.z = a.z + h.z * dd + b.z;
    o.w = a.w + h.w * dd + b.w;
    ((float4*)out)[g] = o;
}

// ---------------- orchestration ----------------

extern "C" void kernel_launch(void* const* d_in, const int* in_sizes, int n_in,
                              void* d_out, int out_size, void* d_ws, size_t ws_size,
                              hipStream_t stream) {
    const float* x  = (const float*)d_in[0];
    const int*   ei = (const int*)d_in[1];
    const float* W1 = (const float*)d_in[2];
    const float* b1 = (const float*)d_in[3];
    const float* W2 = (const float*)d_in[4];
    const float* b2 = (const float*)d_in[5];
    const float* W3 = (const float*)d_in[6];
    const float* b3 = (const float*)d_in[7];

    const int N = in_sizes[0] / IN_F;     // 100000
    const int E = in_sizes[1] / 2;        // 1600000
    const int* srcI = ei;                 // edge_index[0]
    const int* dstI = ei + E;             // edge_index[1]

    // scratch layout (256B aligned slabs)
    char* ws = (char*)d_ws;
    size_t off = 0;
    auto take = [&](size_t bytes) -> char* {
        char* p = ws + off;
        off += (bytes + 255) & ~(size_t)255;
        return p;
    };
    float*  dinv  = (float*)take((size_t)N * 4);
    __bf16* hbfA  = (__bf16*)take((size_t)N * HID_F * 2);
    __bf16* hbfB  = (__bf16*)take((size_t)N * HID_F * 2);
    float*  hlin  = (float*)take((size_t)N * HID_F * 4);
    float*  agg   = (float*)take((size_t)N * HID_F * 4);
    __bf16* w1p   = (__bf16*)take((size_t)IN_F  * HID_F * 2);
    __bf16* w2p   = (__bf16*)take((size_t)HID_F * HID_F * 2);
    __bf16* w3p   = (__bf16*)take((size_t)HID_F * OUT_F * 2);
    (void)ws_size;

    const int TPB = 256;
    const int wavesPerBlk = TPB / 32;
    auto cdiv = [](int a, int b) { return (a + b - 1) / b; };

    // --- normalization: deg = 1 (self loop) + count(dst); dinv = rsqrt(deg)
    k_fill_f32<<<cdiv(N, TPB), TPB, 0, stream>>>(dinv, N, 1.0f);
    k_deg_count<<<cdiv(E, TPB), TPB, 0, stream>>>(dstI, dinv, E);
    k_rsqrt<<<cdiv(N, TPB), TPB, 0, stream>>>(dinv, N);

    // --- bf16 conversion of x; weights packed into WMMA B layout
    k_cvt_f32_bf16<<<cdiv(N * IN_F, TPB), TPB, 0, stream>>>(x, hbfA, N * IN_F);
    k_pack_w<<<cdiv(IN_F * HID_F, TPB), TPB, 0, stream>>>(W1, w1p, IN_F, HID_F);
    k_pack_w<<<cdiv(HID_F * HID_F, TPB), TPB, 0, stream>>>(W2, w2p, HID_F, HID_F);
    k_pack_w<<<cdiv(HID_F * OUT_F, TPB), TPB, 0, stream>>>(W3, w3p, HID_F, OUT_F);

    const int gemmBlks = cdiv(N / 16, wavesPerBlk);   // one wave per 16-row stripe
    const int scatterBlks = cdiv(E, wavesPerBlk);

    // ---------- layer 1: x(bf16) @ W1 -> aggregate -> relu -> hbfB ----------
    k_gemm_wmma_bf16<IN_F, HID_F><<<gemmBlks, TPB, 0, stream>>>(hbfA, w1p, hlin, N);
    k_zero_f32<<<cdiv(N * HID_F, TPB), TPB, 0, stream>>>(agg, N * HID_F);
    k_scatter<<<scatterBlks, TPB, 0, stream>>>(hlin, srcI, dstI, dinv, agg, E, HID_F);
    k_finalize_bf16<<<cdiv(N * HID_F / 4, TPB), TPB, 0, stream>>>(
        agg, hlin, dinv, b1, hbfB, N, HID_F);

    // ---------- layer 2 ----------
    k_gemm_wmma_bf16<HID_F, HID_F><<<gemmBlks, TPB, 0, stream>>>(hbfB, w2p, hlin, N);
    k_zero_f32<<<cdiv(N * HID_F, TPB), TPB, 0, stream>>>(agg, N * HID_F);
    k_scatter<<<scatterBlks, TPB, 0, stream>>>(hlin, srcI, dstI, dinv, agg, E, HID_F);
    k_finalize_bf16<<<cdiv(N * HID_F / 4, TPB), TPB, 0, stream>>>(
        agg, hlin, dinv, b2, hbfA, N, HID_F);

    // ---------- layer 3 (no activation, f32 out) ----------
    k_gemm_wmma_bf16<HID_F, OUT_F><<<gemmBlks, TPB, 0, stream>>>(hbfA, w3p, hlin, N);
    k_zero_f32<<<cdiv(N * OUT_F, TPB), TPB, 0, stream>>>(agg, N * OUT_F);
    k_scatter<<<scatterBlks, TPB, 0, stream>>>(hlin, srcI, dstI, dinv, agg, E, OUT_F);
    k_finalize_f32<<<cdiv(N * OUT_F / 4, TPB), TPB, 0, stream>>>(
        agg, hlin, dinv, b3, (float*)d_out, N, OUT_F);
}
